// AuxModelImageRotation_19988777795713
// MI455X (gfx1250) — matrix operand order repaired and checked by reading, not compiled, not run on hardware
//
#include <hip/hip_runtime.h>
#include <hip/hip_bf16.h>

// ---------------------------------------------------------------------------
// MI455X (gfx1250) implementation — v3.
//
//   1) rotate: obs f32 -> rotated f32 (d_out) ; labels (d_out)
//   2) weight prep: conv1 OIHW->f16 copy; conv2/3 OIHW -> O,KH,KW,IC f16;
//      fc1_w columns permuted NCHW-flatten -> NHWC-flatten, f16
//   3) implicit-GEMM with v_wmma_f32_16x16x32_f16:
//        - 16(M) x 16*NT(N) strip per wave (NT=8 for convs, 2 for fc1)
//        - A tile: 2 x b128 loads, amortized over NT WMMAs
//        - B tiles: channels-last => 2 x b128 contiguous loads per column
//          (conv1 reads f32 NCHW rotated; compiler vectorizes to b128+cvt_pk)
//        - 8 independent WMMAs per K-step hide the WMMA->VALU hazard window
//        - bias+ReLU epilogue packed into one b128 store per tile
//   4) fc2 -> logits f32.
//
// GEMM dims: conv1 M=32 N=246016 K=192 | conv2 M=64 N=50176 K=512
//            conv3 M=64 N=36864  K=576 | fc1 M=512 N=256 K=9216
// All N % (16*NT) == 0, K % 32 == 0, OC % 16 == 0.
// ---------------------------------------------------------------------------

typedef __attribute__((ext_vector_type(16))) _Float16 v16h;
typedef __attribute__((ext_vector_type(8)))  _Float16 v8h;
typedef __attribute__((ext_vector_type(8)))  float    v8f;

// ------------------------------- rotation ----------------------------------
__global__ void rotate_kernel(const float* __restrict__ obs,
                              float* __restrict__ rot) {
  const int HW  = 128 * 128;
  const int per = 64 * 3 * HW;
  const int total = 4 * per;
  for (int o = blockIdx.x * blockDim.x + threadIdx.x; o < total;
       o += gridDim.x * blockDim.x) {
    int a   = o / per;
    int rem = o - a * per;
    int bc  = rem / HW;
    int p   = rem - bc * HW;
    int i = p >> 7, j = p & 127;
    int si, sj;
    if      (a == 0) { si = i;        sj = j;        }
    else if (a == 1) { si = j;        sj = 127 - i;  }
    else if (a == 2) { si = 127 - i;  sj = 127 - j;  }
    else             { si = 127 - j;  sj = i;        }
    rot[o] = obs[(size_t)bc * HW + si * 128 + sj];
  }
}

__global__ void labels_kernel(float* __restrict__ lab) {
  int i = blockIdx.x * blockDim.x + threadIdx.x;
  if (i < 256) lab[i] = (float)(i >> 6);
}

// ----------------------------- weight prep ---------------------------------
__global__ void f32_to_f16_kernel(const float* __restrict__ src,
                                  _Float16* __restrict__ dst, int n) {
  for (int i = blockIdx.x * blockDim.x + threadIdx.x; i < n;
       i += gridDim.x * blockDim.x)
    dst[i] = (_Float16)src[i];
}

// OIHW f32 -> [O][KH][KW][IC] f16 (K-order becomes (kh,kw,ic))
__global__ void wperm_kernel(const float* __restrict__ src,
                             _Float16* __restrict__ dst,
                             int OCn, int ICn, int KHn, int KWn) {
  int total = OCn * ICn * KHn * KWn;
  for (int i = blockIdx.x * blockDim.x + threadIdx.x; i < total;
       i += gridDim.x * blockDim.x) {
    int o  = i / (KHn * KWn * ICn);
    int r  = i - o * (KHn * KWn * ICn);
    int kh = r / (KWn * ICn);
    int r2 = r - kh * (KWn * ICn);
    int kw = r2 / ICn;
    int ic = r2 - kw * ICn;
    dst[i] = (_Float16)src[((o * ICn + ic) * KHn + kh) * KWn + kw];
  }
}

// fc1_w [512][9216] with cols (oc,oy,ox) -> cols (oy,ox,oc) to match NHWC h3
__global__ void fc1perm_kernel(const float* __restrict__ src,
                               _Float16* __restrict__ dst) {
  const int total = 512 * 9216;
  for (int i = blockIdx.x * blockDim.x + threadIdx.x; i < total;
       i += gridDim.x * blockDim.x) {
    int m  = i / 9216;
    int r  = i - m * 9216;
    int p  = r >> 6;       // spatial 0..143
    int oc = r & 63;       // channel 0..63
    dst[i] = (_Float16)src[m * 9216 + oc * 144 + p];
  }
}

// --------- conv1: gather-B implicit GEMM (f32 NCHW in, f16 NHWC out) -------
template <int NT>
__global__ __launch_bounds__(32)
void conv1_wmma_kernel(const float* __restrict__ inp,      // [B,3,128,128] f32
                       const _Float16* __restrict__ wgt,   // [32][192] (ic,kh,kw)
                       const float* __restrict__ bias,
                       _Float16* __restrict__ outp) {      // [B,31,31,32] f16
  constexpr int IC = 3, IH = 128, IW = 128, OC = 32, KH = 8, KW = 8, ST = 4;
  constexpr int OH = 31, OW = 31, K = 192;
  const int lane = threadIdx.x, hl = lane >> 4, lm = lane & 15;

  const int m = (int)blockIdx.y * 16 + lm;
  const _Float16* wrow = wgt + (size_t)m * K;
  __builtin_prefetch(wrow, 0, 3);

  const int ocb = (int)blockIdx.y * 16 + hl * 8;
  float bs[8];
#pragma unroll
  for (int r = 0; r < 8; ++r) bs[r] = bias[ocb + r];

  size_t nbase[NT], obase[NT];
#pragma unroll
  for (int s = 0; s < NT; ++s) {
    int n  = (int)blockIdx.x * (16 * NT) + s * 16 + lm;
    int b  = n / (OH * OW);
    int p  = n - b * (OH * OW);
    int oy = p / OW, ox = p - oy * OW;
    nbase[s] = (size_t)b * IC * IH * IW + (size_t)(oy * ST) * IW + ox * ST;
    obase[s] = (((size_t)b * OH + oy) * OW + ox) * OC + ocb;
  }

  v8f acc[NT];
#pragma unroll
  for (int s = 0; s < NT; ++s) acc[s] = (v8f){};

  for (int k0 = 0; k0 < K; k0 += 32) {
    v8h a0 = *(const v8h*)(wrow + k0 + hl * 16);
    v8h a1 = *(const v8h*)(wrow + k0 + hl * 16 + 8);
    v16h A;
#pragma unroll
    for (int j = 0; j < 8; ++j) { A[j] = a0[j]; A[j + 8] = a1[j]; }

    // per-k NCHW offsets (ic,kh,kw), shared by all NT columns
    int off[16];
#pragma unroll
    for (int j = 0; j < 16; ++j) {
      int k  = k0 + hl * 16 + j;
      int ic = k >> 6;
      int rr = k & 63;
      int kh = rr >> 3, kw = rr & 7;
      off[j] = ic * (IH * IW) + kh * IW + kw;
    }

    v16h Bt[NT];
#pragma unroll
    for (int s = 0; s < NT; ++s)
#pragma unroll
      for (int j = 0; j < 16; ++j)
        Bt[s][j] = (_Float16)inp[nbase[s] + off[j]];

#pragma unroll
    for (int s = 0; s < NT; ++s)
      acc[s] = __builtin_amdgcn_wmma_f32_16x16x32_f16(
          false, A, false, Bt[s], (short)0, acc[s], false, false);
  }

#pragma unroll
  for (int s = 0; s < NT; ++s) {
    v8h o;
#pragma unroll
    for (int r = 0; r < 8; ++r) {
      float v = acc[s][r] + bs[r];
      o[r] = (_Float16)(v > 0.0f ? v : 0.0f);
    }
    *(v8h*)(outp + obase[s]) = o;
  }
}

// ------- channels-last implicit GEMM: conv2 / conv3 / fc1 (f16 in/out) -----
// Requires IC % 16 == 0 so each lane's 16 K-elements are contiguous.
template <int NT, int IC, int IH, int IW, int OC, int KH, int KW, int ST,
          int OH, int OW>
__global__ __launch_bounds__(32)
void conv_cl_wmma_kernel(const _Float16* __restrict__ inp,   // [B,IH,IW,IC]
                         const _Float16* __restrict__ wgt,   // [OC][KH][KW][IC]
                         const float* __restrict__ bias,
                         _Float16* __restrict__ outp) {      // [B,OH,OW,OC]
  constexpr int K = IC * KH * KW;
  static_assert(K % 32 == 0 && IC % 16 == 0 && OC % 16 == 0, "tile mismatch");
  const int lane = threadIdx.x, hl = lane >> 4, lm = lane & 15;

  const int m = (int)blockIdx.y * 16 + lm;
  const _Float16* wrow = wgt + (size_t)m * K;
  __builtin_prefetch(wrow, 0, 3);

  const int ocb = (int)blockIdx.y * 16 + hl * 8;
  float bs[8];
#pragma unroll
  for (int r = 0; r < 8; ++r) bs[r] = bias[ocb + r];

  size_t nbase[NT], obase[NT];
#pragma unroll
  for (int s = 0; s < NT; ++s) {
    int n  = (int)blockIdx.x * (16 * NT) + s * 16 + lm;
    int b  = n / (OH * OW);
    int p  = n - b * (OH * OW);
    int oy = p / OW, ox = p - oy * OW;
    nbase[s] = (((size_t)b * IH + oy * ST) * IW + ox * ST) * IC;
    obase[s] = (((size_t)b * OH + oy) * OW + ox) * OC + ocb;
  }

  v8f acc[NT];
#pragma unroll
  for (int s = 0; s < NT; ++s) acc[s] = (v8f){};

  for (int k0 = 0; k0 < K; k0 += 32) {
    v8h a0 = *(const v8h*)(wrow + k0 + hl * 16);
    v8h a1 = *(const v8h*)(wrow + k0 + hl * 16 + 8);
    v16h A;
#pragma unroll
    for (int j = 0; j < 8; ++j) { A[j] = a0[j]; A[j + 8] = a1[j]; }

    // lane's 16 contiguous K elements: k-order (kh,kw,ic), 16-aligned
    const int kk   = k0 + hl * 16;
    const int kpix = kk / IC;                 // kh*KW + kw
    const int icb  = kk - kpix * IC;
    const int kh   = kpix / KW, kw = kpix - kh * KW;
    const size_t off = ((size_t)kh * IW + kw) * IC + icb;

    v16h Bt[NT];
#pragma unroll
    for (int s = 0; s < NT; ++s) {
      v8h lo = *(const v8h*)(inp + nbase[s] + off);
      v8h hi = *(const v8h*)(inp + nbase[s] + off + 8);
#pragma unroll
      for (int j = 0; j < 8; ++j) { Bt[s][j] = lo[j]; Bt[s][j + 8] = hi[j]; }
    }

#pragma unroll
    for (int s = 0; s < NT; ++s)
      acc[s] = __builtin_amdgcn_wmma_f32_16x16x32_f16(
          false, A, false, Bt[s], (short)0, acc[s], false, false);
  }

#pragma unroll
  for (int s = 0; s < NT; ++s) {
    v8h o;
#pragma unroll
    for (int r = 0; r < 8; ++r) {
      float v = acc[s][r] + bs[r];
      o[r] = (_Float16)(v > 0.0f ? v : 0.0f);
    }
    *(v8h*)(outp + obase[s]) = o;
  }
}

// ------------------------------ fc2 (4 logits) -----------------------------
__global__ void fc2_kernel(const _Float16* __restrict__ h,
                           const float* __restrict__ w,
                           const float* __restrict__ bias,
                           float* __restrict__ out) {
  int t = blockIdx.x * blockDim.x + threadIdx.x;
  if (t >= 256 * 4) return;
  int n = t >> 2, c = t & 3;
  float acc = bias[c];
  const _Float16* hr = h + n * 512;
  const float*    wr = w + c * 512;
#pragma unroll 8
  for (int k = 0; k < 512; ++k) acc += (float)hr[k] * wr[k];
  out[t] = acc;
}

// ------------------------------- launcher ----------------------------------
extern "C" void kernel_launch(void* const* d_in, const int* in_sizes, int n_in,
                              void* d_out, int out_size, void* d_ws,
                              size_t ws_size, hipStream_t stream) {
  (void)in_sizes; (void)n_in; (void)out_size; (void)ws_size;

  const float* obs  = (const float*)d_in[0];
  const float* w1   = (const float*)d_in[1];
  const float* b1   = (const float*)d_in[2];
  const float* w2   = (const float*)d_in[3];
  const float* b2   = (const float*)d_in[4];
  const float* w3   = (const float*)d_in[5];
  const float* b3   = (const float*)d_in[6];
  const float* fc1w = (const float*)d_in[7];
  const float* fc1b = (const float*)d_in[8];
  const float* fc2w = (const float*)d_in[9];
  const float* fc2b = (const float*)d_in[10];

  float* out     = (float*)d_out;
  float* logits  = out;            // [256,4]
  float* labels  = out + 1024;     // [256]
  float* rotated = out + 1280;     // [256,3,128,128] f32 NCHW

  char* ws = (char*)d_ws;
  _Float16* w1h   = (_Float16*)(ws + 0);          // [32][192]
  _Float16* w2h   = (_Float16*)(ws + 12288);      // [64][4][4][32]
  _Float16* w3h   = (_Float16*)(ws + 77824);      // [64][3][3][64]
  _Float16* fc1wh = (_Float16*)(ws + 151552);     // [512][9216] NHWC cols
  _Float16* h1    = (_Float16*)(ws + 9588736);    // [256,31,31,32]
  _Float16* h2    = (_Float16*)(ws + 25333760);   // [256,14,14,64]
  _Float16* h3    = (_Float16*)(ws + 31756288);   // [256,12,12,64]
  _Float16* fc1o  = (_Float16*)(ws + 36474880);   // [256,512]

  // 1) rotation + labels
  rotate_kernel<<<8192, 256, 0, stream>>>(obs, rotated);
  labels_kernel<<<1, 256, 0, stream>>>(labels);

  // 2) weight prep
  f32_to_f16_kernel<<<24, 256, 0, stream>>>(w1, w1h, 32 * 192);
  wperm_kernel<<<128, 256, 0, stream>>>(w2, w2h, 64, 32, 4, 4);
  wperm_kernel<<<144, 256, 0, stream>>>(w3, w3h, 64, 64, 3, 3);
  fc1perm_kernel<<<4096, 256, 0, stream>>>(fc1w, fc1wh);

  // 3) WMMA implicit GEMM
  conv1_wmma_kernel<8>
      <<<dim3(246016 / 128, 2), 32, 0, stream>>>(rotated, w1h, b1, h1);
  conv_cl_wmma_kernel<8, 32, 31, 31, 64, 4, 4, 2, 14, 14>
      <<<dim3(50176 / 128, 4), 32, 0, stream>>>(h1, w2h, b2, h2);
  conv_cl_wmma_kernel<8, 64, 14, 14, 64, 3, 3, 1, 12, 12>
      <<<dim3(36864 / 128, 4), 32, 0, stream>>>(h2, w3h, b3, h3);
  conv_cl_wmma_kernel<2, 9216, 1, 1, 512, 1, 1, 1, 1, 1>
      <<<dim3(256 / 32, 32), 32, 0, stream>>>(h3, fc1wh, fc1b, fc1o);

  // 4) fc2 -> logits
  fc2_kernel<<<8, 128, 0, stream>>>(fc1o, fc2w, fc2b, logits);
}
